// SelfAttention_9577777070305
// MI455X (gfx1250) — matrix operand order repaired and checked by reading, not compiled, not run on hardware
//
#include <hip/hip_runtime.h>

typedef __attribute__((ext_vector_type(16))) _Float16 v16h;
typedef __attribute__((ext_vector_type(8)))  _Float16 v8h;
typedef __attribute__((ext_vector_type(8)))  float    v8f;
typedef __attribute__((ext_vector_type(4)))  int      v4i;

typedef __attribute__((address_space(1))) v4i gv4i;  // global 16B chunk
typedef __attribute__((address_space(3))) v4i lv4i;  // LDS 16B chunk

#define BSZ  32
#define SDIM 1024
#define DDIM 1024

#if __has_builtin(__builtin_amdgcn_global_load_async_to_lds_b128) && \
    __has_builtin(__builtin_amdgcn_s_wait_asynccnt)
#define HAS_ASYNC_LDS 1
#else
#define HAS_ASYNC_LDS 0
#endif

// 16-byte global -> LDS copy. Async DMA (bypasses VGPRs, ASYNCcnt) when the
// toolchain exposes the gfx1250 builtin; synchronous v8h bounce otherwise.
__device__ __forceinline__ void cp16_g2l(const _Float16* __restrict__ g,
                                         _Float16* __restrict__ l) {
#if HAS_ASYNC_LDS
  __builtin_amdgcn_global_load_async_to_lds_b128((gv4i*)g, (lv4i*)l, 0, 0);
#else
  *(v8h*)l = *(const v8h*)g;
#endif
}

__device__ __forceinline__ void wait_g2l() {
#if HAS_ASYNC_LDS
  __builtin_amdgcn_s_wait_asynccnt(0);
#endif
}

// ---------------------------------------------------------------------------
// Kernel 1: x + positional_encoding, transpose [B,S,D] -> [B,D,S], cast f16
// ---------------------------------------------------------------------------
__global__ __launch_bounds__(256) void pe_add_transpose(
    const float* __restrict__ x, _Float16* __restrict__ xt) {
  __shared__ _Float16 tile[32][33];
  const int b  = blockIdx.z;
  const int s0 = blockIdx.y * 32;
  const int d0 = blockIdx.x * 32;
  const int t  = threadIdx.x;
  const float invD = 1.0f / (float)DDIM;
#pragma unroll
  for (int i = 0; i < 4; ++i) {
    int e  = t + 256 * i;
    int ts = e >> 5, td = e & 31;
    int s = s0 + ts, d = d0 + td;
    float xv   = x[(size_t)b * SDIM * DDIM + (size_t)s * DDIM + d];
    float even = (float)((d >> 1) << 1);
    float rate = __powf(10000.0f, -even * invD);
    float ang  = (float)s * rate;
    float pe   = (d & 1) ? __cosf(ang) : __sinf(ang);
    tile[ts][td] = (_Float16)(xv + pe);
  }
  __syncthreads();
#pragma unroll
  for (int i = 0; i < 4; ++i) {
    int e  = t + 256 * i;
    int od = e >> 5, os = e & 31;
    xt[(size_t)b * DDIM * SDIM + (size_t)(d0 + od) * SDIM + (s0 + os)] =
        tile[os][od];
  }
}

// ---------------------------------------------------------------------------
// Kernel 2: f32 -> f16 convert (weights)
// ---------------------------------------------------------------------------
__global__ __launch_bounds__(256) void convert_f32_f16(
    const float* __restrict__ s, _Float16* __restrict__ d, int n) {
  int i = blockIdx.x * 256 + threadIdx.x;
  if (i < n) d[i] = (_Float16)s[i];
}

// ---------------------------------------------------------------------------
// Kernel 3: f16 batched transpose src[b][m][n] -> dst[b][n][m]
// ---------------------------------------------------------------------------
__global__ __launch_bounds__(256) void transpose_h(
    const _Float16* __restrict__ src, _Float16* __restrict__ dst,
    int M, int N) {
  __shared__ _Float16 tile[32][33];
  const int b  = blockIdx.z;
  const int m0 = blockIdx.y * 32;
  const int n0 = blockIdx.x * 32;
  const int t  = threadIdx.x;
#pragma unroll
  for (int i = 0; i < 4; ++i) {
    int e  = t + 256 * i;
    int tm = e >> 5, tn = e & 31;
    tile[tm][tn] =
        src[(size_t)b * M * N + (size_t)(m0 + tm) * N + (n0 + tn)];
  }
  __syncthreads();
#pragma unroll
  for (int i = 0; i < 4; ++i) {
    int e  = t + 256 * i;
    int on = e >> 5, om = e & 31;
    dst[(size_t)b * M * N + (size_t)(n0 + on) * M + (m0 + om)] =
        tile[om][on];
  }
}

// ---------------------------------------------------------------------------
// Kernel 4: WMMA GEMM  C[b][m][n] = scale * sum_k A[b][m][k]*B[b][n][k] + bias[n]
// A, B both K-contiguous f16. Block tile 128x128, K panels of 32, double-
// buffered LDS with async global->LDS DMA overlapped against WMMA.
// 8 waves (2x4); each wave: 64x32 tile = 8x v_wmma_f32_16x16x32_f16 per panel.
// ---------------------------------------------------------------------------
template <bool OUT_F16>
__global__ __launch_bounds__(256) void gemm_wmma(
    const _Float16* __restrict__ A, size_t aStride,
    const _Float16* __restrict__ Bm, size_t bStride,
    const float* __restrict__ bias,
    void* __restrict__ C, size_t cStride,
    int M, int N, int K, float scale) {
  __shared__ _Float16 lA[2][128 * 40];  // 32 halves + 8 pad per row (16B align)
  __shared__ _Float16 lB[2][128 * 40];

  const int t    = threadIdx.x;
  const int lane = t & 31;
  const int w    = t >> 5;
  const int wm   = (w >> 2) * 64;  // wave M offset in block tile (0 / 64)
  const int wn   = (w & 3) * 32;   // wave N offset (0,32,64,96)
  const int lr   = lane & 15;
  const int hh   = lane >> 4;      // half-wave selector
  const int tm   = blockIdx.y * 128;
  const int tn   = blockIdx.x * 128;
  const size_t aOff = (size_t)blockIdx.z * aStride;
  const size_t bOff = (size_t)blockIdx.z * bStride;
  const size_t cOff = (size_t)blockIdx.z * cStride;

  v8f acc[4][2];
#pragma unroll
  for (int mi = 0; mi < 4; ++mi)
#pragma unroll
    for (int ni = 0; ni < 2; ++ni)
#pragma unroll
      for (int e = 0; e < 8; ++e) acc[mi][ni][e] = 0.0f;

  // issue async copies for one 128x32 A panel + 128x32 B panel
  auto stage = [&](int bi, int k0) {
#pragma unroll
    for (int i = 0; i < 2; ++i) {
      int c   = t * 2 + i;          // 0..511
      int row = c >> 2;             // 0..127
      int kc  = (c & 3) * 8;        // 0,8,16,24
      cp16_g2l(A + aOff + (size_t)(tm + row) * K + (k0 + kc),
               &lA[bi][row * 40 + kc]);
      cp16_g2l(Bm + bOff + (size_t)(tn + row) * K + (k0 + kc),
               &lB[bi][row * 40 + kc]);
    }
  };

  // fragments per CDNA5 16-bit A/B VGPR layouts + 8 WMMAs
  auto compute = [&](int bi) {
    v16h af[4];
#pragma unroll
    for (int mi = 0; mi < 4; ++mi) {
      const _Float16* p = &lA[bi][(wm + mi * 16 + lr) * 40];
      // lanes 0-15: K 0-7 then 16-23; lanes 16-31: K 8-15 then 24-31
      v8h lo = *(const v8h*)&p[hh * 8];
      v8h hi = *(const v8h*)&p[hh * 8 + 16];
      af[mi] = __builtin_shufflevector(lo, hi, 0, 1, 2, 3, 4, 5, 6, 7, 8, 9,
                                       10, 11, 12, 13, 14, 15);
    }
    v16h bfr[2];
#pragma unroll
    for (int ni = 0; ni < 2; ++ni) {
      const _Float16* p = &lB[bi][(wn + ni * 16 + lr) * 40];
      // lanes 0-15: K 0-7 then 16-23; lanes 16-31: K 8-15 then 24-31
      v8h lo = *(const v8h*)&p[hh * 8];
      v8h hi = *(const v8h*)&p[hh * 8 + 16];
      bfr[ni] = __builtin_shufflevector(lo, hi, 0, 1, 2, 3, 4, 5, 6, 7, 8, 9,
                                        10, 11, 12, 13, 14, 15);
    }
#pragma unroll
    for (int mi = 0; mi < 4; ++mi)
#pragma unroll
      for (int ni = 0; ni < 2; ++ni)
        acc[mi][ni] = __builtin_amdgcn_wmma_f32_16x16x32_f16(
            false, af[mi], false, bfr[ni], (short)0, acc[mi][ni], false,
            false);
  };

  // prologue: fill buffer 0
  stage(0, 0);
  wait_g2l();
  __syncthreads();

  int buf = 0;
  for (int k0 = 32; k0 < K; k0 += 32) {
    stage(buf ^ 1, k0);   // DMA next panel into the other buffer
    compute(buf);         // WMMA on current panel (overlaps the DMA)
    wait_g2l();
    __syncthreads();
    buf ^= 1;
  }
  compute(buf);

  // ---- epilogue: C layout = VGPR e: lanes 0-15 -> M=e, lanes 16-31 -> M=e+8
#pragma unroll
  for (int mi = 0; mi < 4; ++mi) {
#pragma unroll
    for (int ni = 0; ni < 2; ++ni) {
      int col  = tn + wn + ni * 16 + lr;
      float bv = bias ? bias[col] : 0.0f;
#pragma unroll
      for (int e = 0; e < 8; ++e) {
        int row = tm + wm + mi * 16 + hh * 8 + e;
        float r = acc[mi][ni][e] * scale + bv;
        if (OUT_F16)
          ((_Float16*)C)[cOff + (size_t)row * N + col] = (_Float16)r;
        else
          ((float*)C)[cOff + (size_t)row * N + col] = r;
      }
    }
  }
}

// ---------------------------------------------------------------------------
// Kernel 5: row softmax over 1024 elems; f32 in-place + f16 copy for next GEMM
// ---------------------------------------------------------------------------
__global__ __launch_bounds__(256) void softmax_rows(
    float* __restrict__ scores, _Float16* __restrict__ attn_h) {
  const size_t row = blockIdx.x;
  float* p = scores + row * 1024;
  const int t = threadIdx.x;
  __shared__ float red[8];

  float vals[4];
  float mx = -3.402823466e38f;
#pragma unroll
  for (int j = 0; j < 4; ++j) {
    vals[j] = p[t + 256 * j];
    mx = fmaxf(mx, vals[j]);
  }
#pragma unroll
  for (int off = 16; off > 0; off >>= 1)
    mx = fmaxf(mx, __shfl_xor(mx, off, 32));
  if ((t & 31) == 0) red[t >> 5] = mx;
  __syncthreads();
  mx = red[0];
#pragma unroll
  for (int i = 1; i < 8; ++i) mx = fmaxf(mx, red[i]);

  float sum = 0.0f;
#pragma unroll
  for (int j = 0; j < 4; ++j) {
    vals[j] = __expf(vals[j] - mx);
    sum += vals[j];
  }
#pragma unroll
  for (int off = 16; off > 0; off >>= 1) sum += __shfl_xor(sum, off, 32);
  __syncthreads();
  if ((t & 31) == 0) red[t >> 5] = sum;
  __syncthreads();
  sum = red[0];
#pragma unroll
  for (int i = 1; i < 8; ++i) sum += red[i];
  float inv = 1.0f / sum;

#pragma unroll
  for (int j = 0; j < 4; ++j) {
    float a = vals[j] * inv;
    p[t + 256 * j] = a;
    attn_h[row * 1024 + t + 256 * j] = (_Float16)a;
  }
}

// ---------------------------------------------------------------------------
extern "C" void kernel_launch(void* const* d_in, const int* in_sizes, int n_in,
                              void* d_out, int out_size, void* d_ws,
                              size_t ws_size, hipStream_t stream) {
  (void)in_sizes; (void)n_in; (void)out_size; (void)ws_size;
  const float* x  = (const float*)d_in[0];
  const float* Wq = (const float*)d_in[1];
  const float* bq = (const float*)d_in[2];
  const float* Wk = (const float*)d_in[3];
  const float* bk = (const float*)d_in[4];
  const float* Wv = (const float*)d_in[5];
  const float* bv = (const float*)d_in[6];

  float* attn_out     = (float*)d_out;                           // [B,D,D]
  float* attended_out = attn_out + (size_t)BSZ * DDIM * DDIM;    // [B,D,S]

  const size_t BDS = (size_t)BSZ * DDIM * SDIM;
  _Float16* base   = (_Float16*)d_ws;
  _Float16* xt_h   = base;             // [B,D,S]
  _Float16* q_h    = base + BDS;       // [B,D,S]
  _Float16* k_h    = base + 2 * BDS;   // [B,D,S]
  _Float16* v_h    = base + 3 * BDS;   // [B,D,S]
  _Float16* vT_h   = base + 4 * BDS;   // [B,S,D]  (e K-contiguous)
  _Float16* attn_h = base + 5 * BDS;   // [B,D,D]
  _Float16* wq_h   = base + 6 * BDS;
  _Float16* wk_h   = wq_h + (size_t)SDIM * SDIM;
  _Float16* wv_h   = wk_h + (size_t)SDIM * SDIM;

  const size_t DSst = (size_t)DDIM * SDIM;
  const size_t DDst = (size_t)DDIM * DDIM;

  // 1) x + PE, transpose, cast f16
  dim3 pgrid(DDIM / 32, SDIM / 32, BSZ);
  pe_add_transpose<<<pgrid, 256, 0, stream>>>(x, xt_h);

  // 2) weights to f16
  const int nW = SDIM * SDIM;
  convert_f32_f16<<<(nW + 255) / 256, 256, 0, stream>>>(Wq, wq_h, nW);
  convert_f32_f16<<<(nW + 255) / 256, 256, 0, stream>>>(Wk, wk_h, nW);
  convert_f32_f16<<<(nW + 255) / 256, 256, 0, stream>>>(Wv, wv_h, nW);

  // 3) Q/K/V projections: M=D, N=S, K=S; B = W (K-major), batched A
  dim3 ggrid(SDIM / 128, DDIM / 128, BSZ);
  gemm_wmma<true><<<ggrid, 256, 0, stream>>>(xt_h, DSst, wq_h, 0, bq, q_h,
                                             DSst, DDIM, SDIM, SDIM, 1.0f);
  gemm_wmma<true><<<ggrid, 256, 0, stream>>>(xt_h, DSst, wk_h, 0, bk, k_h,
                                             DSst, DDIM, SDIM, SDIM, 1.0f);
  gemm_wmma<true><<<ggrid, 256, 0, stream>>>(xt_h, DSst, wv_h, 0, bv, v_h,
                                             DSst, DDIM, SDIM, SDIM, 1.0f);

  // 4) transpose V so its K (=e) dim is contiguous for the attended GEMM
  dim3 vgrid(SDIM / 32, DDIM / 32, BSZ);
  transpose_h<<<vgrid, 256, 0, stream>>>(v_h, vT_h, DDIM, SDIM);

  // 5) scores = Q.K^T / sqrt(S): M=D, N=D, K=S; f32 into attn output region
  dim3 sgrid(DDIM / 128, DDIM / 128, BSZ);
  gemm_wmma<false><<<sgrid, 256, 0, stream>>>(q_h, DSst, k_h, DSst, nullptr,
                                              attn_out, DDst, DDIM, DDIM,
                                              SDIM, 0.03125f);

  // 6) softmax rows (in-place f32 + f16 copy)
  softmax_rows<<<BSZ * DDIM, 256, 0, stream>>>(attn_out, attn_h);

  // 7) attended = attn.V: M=D, N=S, K=D (vT is K-contiguous)
  gemm_wmma<false><<<ggrid, 256, 0, stream>>>(attn_h, DDst, vT_h, DSst,
                                              nullptr, attended_out, DSst,
                                              DDIM, SDIM, DDIM, 1.0f);
}
